// StructuredPriorAttention_31568009625736
// MI455X (gfx1250) — matrix operand order repaired and checked by reading, not compiled, not run on hardware
//
#include <hip/hip_runtime.h>

#ifndef __has_builtin
#define __has_builtin(x) 0
#endif

#if __has_builtin(__builtin_amdgcn_global_load_async_to_lds_b128)
#define HAVE_ASYNC_LDS 1
#else
#define HAVE_ASYNC_LDS 0
#endif

#if __has_builtin(__builtin_amdgcn_s_wait_asynccnt)
#define WAIT_ASYNC(n) __builtin_amdgcn_s_wait_asynccnt(n)
#else
#define WAIT_ASYNC(n) asm volatile("s_wait_asynccnt %0" :: "i"(n) : "memory")
#endif

#if __has_builtin(__builtin_amdgcn_tanhf)
#define FAST_TANH(x) __builtin_amdgcn_tanhf(x)
#else
#define FAST_TANH(x) tanhf(x)
#endif

typedef __attribute__((ext_vector_type(16))) _Float16 v16h;
typedef __attribute__((ext_vector_type(8)))  _Float16 v8h;
typedef __attribute__((ext_vector_type(8)))  float    v8f;
typedef __attribute__((ext_vector_type(4)))  int      v4i;
typedef _Float16 half_t;

// pointer-to-addrspace(1)/(3) int4, as required by the async-LDS builtin
typedef __attribute__((address_space(1))) v4i* gv4i_p;
typedef __attribute__((address_space(3))) v4i* lv4i_p;

#define NUM_HEADS 12
#define HD   64
#define SEQ  1024
#define CDIM 768
#define BATCH 8
#define ROWS (BATCH * SEQ)      // 8192
#define OC3  (3 * CDIM)         // 2304

// scale = D^-0.5 = 1/8 ; gamma = sqrt((1+alpha^2)/2) = sqrt(0.625)
#define GAMMA_F 0.7905694150420949f
#define C1_F (0.75f * 0.125f / GAMMA_F)
#define C2_F (0.25f * 0.125f / GAMMA_F)
#define CLIP_F 5.0f

// ---------------------------------------------------------------------------
// 16-byte global -> LDS copy: async (ASYNCcnt, no VGPR round trip) when the
// gfx1250 builtin exists, else synchronous b128 copy.
// ---------------------------------------------------------------------------
__device__ __forceinline__ void async_cp16(const half_t* gsrc, half_t* ldst) {
#if HAVE_ASYNC_LDS
    __builtin_amdgcn_global_load_async_to_lds_b128(
        (gv4i_p)(void*)(const void*)gsrc, (lv4i_p)(void*)ldst, 0, 0);
#else
    *(v8h*)ldst = *(const v8h*)gsrc;
#endif
}

// ---------------------------------------------------------------------------
// WMMA fragment helpers (v_wmma_f32_16x16x32_f16, wave32)
// A 16x32: lanes 0-15 row M=lane, K 0..7 & 16..23 ; lanes 16-31 K 8..15 & 24..31
// B 32x16: col = lane&15, lanes 0-15 K 0..7 & 16..23 ; lanes 16-31 K 8..15 & 24..31 (source W[col][k])
// C/D 16x16: VGPR g -> row g (lanes 0-15) / g+8 (lanes 16-31), col = lane&15
// ---------------------------------------------------------------------------
__device__ __forceinline__ v16h frag_a_ld(const half_t* base, int stride, int lane) {
    int row = lane & 15;
    int kb  = (lane >> 4) << 3;
    const half_t* p = base + row * stride + kb;
    v8h a0 = *(const v8h*)(p);
    v8h a1 = *(const v8h*)(p + 16);
    v16h r;
#pragma unroll
    for (int i = 0; i < 8; ++i) { r[i] = a0[i]; r[8 + i] = a1[i]; }
    return r;
}

__device__ __forceinline__ v16h frag_b_ld(const half_t* base, int stride, int lane) {
    int col = lane & 15;
    int kb  = (lane >> 4) << 3;
    const half_t* p = base + col * stride + kb;
    v8h b0 = *(const v8h*)(p);
    v8h b1 = *(const v8h*)(p + 16);
    v16h r;
#pragma unroll
    for (int i = 0; i < 8; ++i) { r[i] = b0[i]; r[8 + i] = b1[i]; }
    return r;
}

__device__ __forceinline__ v8f wmma16(v16h a, v16h b, v8f c) {
    return __builtin_amdgcn_wmma_f32_16x16x32_f16(false, a, false, b, (short)0, c,
                                                  false, false);
}

// ---------------------------------------------------------------------------
// Kernel 1: QKV GEMM (f32 in, f16 WMMA, f32 acc) + scaled scatter.
// ---------------------------------------------------------------------------
__global__ __launch_bounds__(128)
void qkv_kernel(const float* __restrict__ x, const float* __restrict__ qkv_w,
                const float* __restrict__ qkv_b,
                half_t* __restrict__ Qp, half_t* __restrict__ Kp,
                half_t* __restrict__ Vt) {
    __shared__ half_t As[64 * 72];
    __shared__ half_t Bs[64 * 72];
    const int tid  = threadIdx.x;
    const int wv   = tid >> 5;
    const int lane = tid & 31;
    const int r0   = blockIdx.x * 64;   // over 8192 rows
    const int o0   = blockIdx.y * 64;   // over 2304 cols
    v8f acc[4] = {};

    for (int kc = 0; kc < CDIM; kc += 64) {
        // phase 1: batch all global loads into registers (overlap latency)
        float4 xb[8], wb4[8];
#pragma unroll
        for (int i = 0; i < 8; ++i) {
            int slot = tid + i * 128;
            int row  = slot >> 4;
            int c4   = (slot & 15) << 2;
            xb[i]  = *(const float4*)(x + (size_t)(r0 + row) * CDIM + kc + c4);
            wb4[i] = *(const float4*)(qkv_w + (size_t)(o0 + row) * CDIM + kc + c4);
        }
        // phase 2: convert + store to LDS
#pragma unroll
        for (int i = 0; i < 8; ++i) {
            int slot = tid + i * 128;
            int row  = slot >> 4;
            int c4   = (slot & 15) << 2;
            half_t* da = &As[row * 72 + c4];
            da[0] = (half_t)xb[i].x; da[1] = (half_t)xb[i].y;
            da[2] = (half_t)xb[i].z; da[3] = (half_t)xb[i].w;
            half_t* db = &Bs[row * 72 + c4];
            db[0] = (half_t)wb4[i].x; db[1] = (half_t)wb4[i].y;
            db[2] = (half_t)wb4[i].z; db[3] = (half_t)wb4[i].w;
        }
        __syncthreads();
#pragma unroll
        for (int kk = 0; kk < 64; kk += 32) {
            v16h a = frag_a_ld(&As[(wv * 16) * 72 + kk], 72, lane);
            v16h bf[4];
#pragma unroll
            for (int t = 0; t < 4; ++t)
                bf[t] = frag_b_ld(&Bs[(t * 16) * 72 + kk], 72, lane);
#pragma unroll
            for (int t = 0; t < 4; ++t) acc[t] = wmma16(a, bf[t], acc[t]);
        }
        __syncthreads();
    }

    const int hl = (lane >> 4) << 3;
#pragma unroll
    for (int t = 0; t < 4; ++t) {
        int o = o0 + t * 16 + (lane & 15);
        float bias = qkv_b[o];
        int which = o / CDIM;
        int h     = (o % CDIM) >> 6;
        int d     = o & 63;
#pragma unroll
        for (int g = 0; g < 8; ++g) {
            int n_glob = r0 + wv * 16 + g + hl;
            int b      = n_glob >> 10;
            int n      = n_glob & (SEQ - 1);
            size_t base = ((size_t)((b * NUM_HEADS + h) * SEQ) + n) * 128;
            float val = acc[t][g] + bias;
            if (which == 0) {            // q
                Qp[base + d]      = (half_t)(val * C1_F);
                Kp[base + 64 + d] = (half_t)val;
            } else if (which == 1) {     // k
                Qp[base + 64 + d] = (half_t)(val * C2_F);
                Kp[base + d]      = (half_t)val;
            } else {                     // v (transposed)
                Vt[((size_t)((b * NUM_HEADS + h) * HD) + d) * SEQ + n] = (half_t)val;
            }
        }
    }
}

// ---------------------------------------------------------------------------
// Issue the 12 async b128 copies for one (K' 64x128, V^T 64x64) tile pair.
// ---------------------------------------------------------------------------
__device__ __forceinline__ void issue_tile(const half_t* Kpb, const half_t* Vtb,
                                           int j, half_t* ksbuf, half_t* vsbuf,
                                           int tid) {
    {
        int row = tid >> 1, seg = (tid & 1) * 64;
        const half_t* src = Kpb + (size_t)(j + row) * 128 + seg;
        half_t* dst = ksbuf + row * 136 + seg;
#pragma unroll
        for (int i = 0; i < 8; ++i) async_cp16(src + i * 8, dst + i * 8);
    }
    {
        int d = tid >> 1, seg = (tid & 1) * 32;
        const half_t* src = Vtb + (size_t)d * SEQ + j + seg;
        half_t* dst = vsbuf + d * 72 + seg;
#pragma unroll
        for (int i = 0; i < 4; ++i) async_cp16(src + i * 8, dst + i * 8);
    }
}

// ---------------------------------------------------------------------------
// Kernel 2: flash attention, double-buffered async tiles, fixed-max softmax.
// tanh clips scores to [-5,5] => softmax can use static max 5: p = exp(s-5).
// Row sums accumulated per-lane, reduced across 16 lanes once at the end.
// ---------------------------------------------------------------------------
__global__ __launch_bounds__(128)
void attn_kernel(const half_t* __restrict__ Qp, const half_t* __restrict__ Kp,
                 const half_t* __restrict__ Vt, half_t* __restrict__ Aout) {
    __shared__ half_t Ks[2][64 * 136];   // K' tiles: 64 rows x 128 (+8 pad)
    __shared__ half_t Vs[2][64 * 72];    // V^T tiles: 64 d-rows x 64 (+8 pad)
    __shared__ half_t Ps[4 * 16 * 72];   // per-wave P staging
    const int tid  = threadIdx.x;
    const int wv   = tid >> 5;
    const int lane = tid & 31;
    const int bh   = blockIdx.x;
    const int b    = bh / NUM_HEADS;
    const int h    = bh % NUM_HEADS;
    const int q0   = blockIdx.y * 64;
    const half_t* Kpb = Kp + (size_t)bh * SEQ * 128;
    const half_t* Vtb = Vt + (size_t)bh * HD * SEQ;

    // Q' A-fragments (16 rows x K=128 -> 4 fragments), kept in registers
    v16h aq[4];
    {
        int row = q0 + wv * 16 + (lane & 15);
        const half_t* qrow = Qp + (size_t)bh * SEQ * 128 + (size_t)row * 128;
        int kb = (lane >> 4) << 3;
#pragma unroll
        for (int kk = 0; kk < 4; ++kk) {
            v8h a0 = *(const v8h*)(qrow + kk * 32 + kb);
            v8h a1 = *(const v8h*)(qrow + kk * 32 + kb + 16);
#pragma unroll
            for (int i = 0; i < 8; ++i) { aq[kk][i] = a0[i]; aq[kk][8 + i] = a1[i]; }
        }
    }

    float lsum[8];
#pragma unroll
    for (int g = 0; g < 8; ++g) lsum[g] = 0.0f;
    v8f o[4] = {};

    issue_tile(Kpb, Vtb, 0, Ks[0], Vs[0], tid);   // prologue: tile 0 in flight

    for (int it = 0; it < SEQ / 64; ++it) {
        const int cur = it & 1;
        if (it + 1 < SEQ / 64) {
            issue_tile(Kpb, Vtb, (it + 1) * 64, Ks[cur ^ 1], Vs[cur ^ 1], tid);
            WAIT_ASYNC(12);   // async loads retire in order: tile `it` landed
        } else {
            WAIT_ASYNC(0);
        }
        __syncthreads();

        // S = Q' @ K'^T   (symmetrized + scaled via doubled-K trick)
        v8f s[4] = {};
#pragma unroll
        for (int kk = 0; kk < 4; ++kk) {
            v16h bf[4];
#pragma unroll
            for (int t = 0; t < 4; ++t)
                bf[t] = frag_b_ld(&Ks[cur][(t * 16) * 136 + kk * 32], 136, lane);
#pragma unroll
            for (int t = 0; t < 4; ++t) s[t] = wmma16(aq[kk], bf[t], s[t]);
        }

        // p = exp(5*tanh(s/5) - 5); accumulate per-lane row partial sums
#pragma unroll
        for (int g = 0; g < 8; ++g) {
#pragma unroll
            for (int t = 0; t < 4; ++t) {
                float v_ = CLIP_F * FAST_TANH(s[t][g] * (1.0f / CLIP_F));
                float p  = __expf(v_ - CLIP_F);
                s[t][g] = p;
                lsum[g] += p;
            }
        }

        // stage P (C-layout f32 -> row-major f16) in this wave's LDS region
        {
            half_t* pw = &Ps[wv * 16 * 72];
            int hl = (lane >> 4) << 3;
            int cl = lane & 15;
#pragma unroll
            for (int t = 0; t < 4; ++t)
#pragma unroll
                for (int g = 0; g < 8; ++g)
                    pw[(g + hl) * 72 + t * 16 + cl] = (half_t)s[t][g];
        }
        // O += P @ V   (same-wave DS ordering: no barrier needed for Ps)
        {
            const half_t* pw = &Ps[wv * 16 * 72];
#pragma unroll
            for (int kk = 0; kk < 2; ++kk) {
                v16h pa = frag_a_ld(pw + kk * 32, 72, lane);
                v16h bv[4];
#pragma unroll
                for (int t = 0; t < 4; ++t)
                    bv[t] = frag_b_ld(&Vs[cur][(t * 16) * 72 + kk * 32], 72, lane);
#pragma unroll
                for (int t = 0; t < 4; ++t) o[t] = wmma16(pa, bv[t], o[t]);
            }
        }
        __syncthreads();   // all waves done reading tile `it` before its buffer reuse
    }

    // deferred row-sum reduction (16-lane groups), normalize, write f16 output
    {
        int hl = (lane >> 4) << 3;
        int cl = lane & 15;
#pragma unroll
        for (int g = 0; g < 8; ++g) {
            float rs = lsum[g];
            rs += __shfl_xor(rs, 1, 32);
            rs += __shfl_xor(rs, 2, 32);
            rs += __shfl_xor(rs, 4, 32);
            rs += __shfl_xor(rs, 8, 32);
            float inv = 1.0f / rs;
            int n = q0 + wv * 16 + g + hl;
            size_t rowbase = ((size_t)(b * SEQ + n)) * CDIM + h * HD;
#pragma unroll
            for (int t = 0; t < 4; ++t)
                Aout[rowbase + t * 16 + cl] = (half_t)(o[t][g] * inv);
        }
    }
}

// ---------------------------------------------------------------------------
// Kernel 3: output projection GEMM + bias (f32 out)
// ---------------------------------------------------------------------------
__global__ __launch_bounds__(128)
void proj_kernel(const half_t* __restrict__ A16, const float* __restrict__ proj_w,
                 const float* __restrict__ proj_b, float* __restrict__ out) {
    __shared__ half_t As[64 * 72];
    __shared__ half_t Bs[64 * 72];
    const int tid  = threadIdx.x;
    const int wv   = tid >> 5;
    const int lane = tid & 31;
    const int r0   = blockIdx.x * 64;
    const int o0   = blockIdx.y * 64;
    v8f acc[4] = {};

    for (int kc = 0; kc < CDIM; kc += 64) {
        // A tile: pure f16 copy -> async path
#pragma unroll
        for (int i = 0; i < 4; ++i) {
            int slot = tid + i * 128;          // 512 b128 slots
            int row  = slot >> 3;
            int g8   = (slot & 7) << 3;
            async_cp16(A16 + (size_t)(r0 + row) * CDIM + kc + g8, &As[row * 72 + g8]);
        }
        // W tile: register-stage f32 loads, then convert+store
        float4 wb4[8];
#pragma unroll
        for (int i = 0; i < 8; ++i) {
            int slot = tid + i * 128;
            int row  = slot >> 4;
            int c4   = (slot & 15) << 2;
            wb4[i] = *(const float4*)(proj_w + (size_t)(o0 + row) * CDIM + kc + c4);
        }
#pragma unroll
        for (int i = 0; i < 8; ++i) {
            int slot = tid + i * 128;
            int row  = slot >> 4;
            int c4   = (slot & 15) << 2;
            half_t* db = &Bs[row * 72 + c4];
            db[0] = (half_t)wb4[i].x; db[1] = (half_t)wb4[i].y;
            db[2] = (half_t)wb4[i].z; db[3] = (half_t)wb4[i].w;
        }
        WAIT_ASYNC(0);
        __syncthreads();
#pragma unroll
        for (int kk = 0; kk < 64; kk += 32) {
            v16h a = frag_a_ld(&As[(wv * 16) * 72 + kk], 72, lane);
            v16h bf[4];
#pragma unroll
            for (int t = 0; t < 4; ++t)
                bf[t] = frag_b_ld(&Bs[(t * 16) * 72 + kk], 72, lane);
#pragma unroll
            for (int t = 0; t < 4; ++t) acc[t] = wmma16(a, bf[t], acc[t]);
        }
        __syncthreads();
    }

    const int hl = (lane >> 4) << 3;
#pragma unroll
    for (int t = 0; t < 4; ++t) {
        int oc = o0 + t * 16 + (lane & 15);
        float bias = proj_b[oc];
#pragma unroll
        for (int g = 0; g < 8; ++g) {
            int r = r0 + wv * 16 + g + hl;
            out[(size_t)r * CDIM + oc] = acc[t][g] + bias;
        }
    }
}

// ---------------------------------------------------------------------------
extern "C" void kernel_launch(void* const* d_in, const int* in_sizes, int n_in,
                              void* d_out, int out_size, void* d_ws, size_t ws_size,
                              hipStream_t stream) {
    (void)in_sizes; (void)n_in; (void)out_size; (void)ws_size;
    const float* x      = (const float*)d_in[0];
    const float* qkv_w  = (const float*)d_in[1];
    const float* qkv_b  = (const float*)d_in[2];
    const float* proj_w = (const float*)d_in[3];
    const float* proj_b = (const float*)d_in[4];
    float* out = (float*)d_out;

    char* ws = (char*)d_ws;
    const size_t qp_sz = (size_t)BATCH * NUM_HEADS * SEQ * 128 * sizeof(half_t); // 24 MB
    const size_t vt_sz = (size_t)BATCH * NUM_HEADS * HD * SEQ * sizeof(half_t);  // 12 MB
    half_t* Qp   = (half_t*)(ws);
    half_t* Kp   = (half_t*)(ws + qp_sz);
    half_t* Vt   = (half_t*)(ws + 2 * qp_sz);
    half_t* Aout = (half_t*)(ws + 2 * qp_sz + vt_sz);

    dim3 blk(128);
    qkv_kernel<<<dim3(ROWS / 64, OC3 / 64), blk, 0, stream>>>(x, qkv_w, qkv_b,
                                                              Qp, Kp, Vt);
    attn_kernel<<<dim3(BATCH * NUM_HEADS, SEQ / 64), blk, 0, stream>>>(Qp, Kp, Vt,
                                                                       Aout);
    proj_kernel<<<dim3(ROWS / 64, CDIM / 64), blk, 0, stream>>>(Aout, proj_w,
                                                                proj_b, out);
}
